// SparseConvNetBackbone_3307124818327
// MI455X (gfx1250) — matrix operand order repaired and checked by reading, not compiled, not run on hardware
//
#include <hip/hip_runtime.h>

typedef unsigned short u16;
typedef __attribute__((ext_vector_type(16))) __bf16 v16bf;
typedef __attribute__((ext_vector_type(8)))  float  v8f;

#define BN_EPS 1e-4f

__device__ __forceinline__ u16 f2bf(float f){
  unsigned int u = __builtin_bit_cast(unsigned int, f);
  u += 0x7FFFu + ((u >> 16) & 1u);           // round-to-nearest-even
  return (u16)(u >> 16);
}
__device__ __forceinline__ float bf2f(u16 h){
  unsigned int u = ((unsigned int)h) << 16;
  return __builtin_bit_cast(float, u);
}

union Frag { v16bf v; u16 u[16]; uint4 q[2]; };

// ---------------------------------------------------------------- utility
__global__ __launch_bounds__(256) void k_zero(float* p, int n){
  int i = blockIdx.x*256 + threadIdx.x;
  if (i < n) p[i] = 0.f;
}

__global__ __launch_bounds__(256) void k_mask(const float* __restrict__ v, float* __restrict__ m, int n){
  int i = blockIdx.x*256 + threadIdx.x;
  if (i < n) m[i] = (fabsf(v[i]) > 0.f) ? 1.f : 0.f;
}

__global__ __launch_bounds__(256) void k_pool_mask(const float* __restrict__ mi, float* __restrict__ mo,
                                                   int lro, int n){
  int i = blockIdx.x*256 + threadIdx.x;
  if (i >= n) return;
  int Ro = 1 << lro, lri = lro + 1;
  int x =  i & (Ro-1);
  int y = (i >> lro) & (Ro-1);
  int z = (i >> (2*lro)) & (Ro-1);
  int b =  i >> (3*lro);
  float m = 0.f;
  int base = ((((b << lri) + 2*z) << lri) + 2*y) << lri;
  base += 2*x;
  for (int dz=0; dz<2; ++dz)
    for (int dy=0; dy<2; ++dy)
      for (int dx=0; dx<2; ++dx)
        m = fmaxf(m, mi[(size_t)base + (dz << (2*lri)) + (dy << lri) + dx]);
  mo[i] = m;
}

// ---------------------------------------------------------------- conv0 (1->32, direct)
__global__ __launch_bounds__(256) void k_conv0(const float* __restrict__ vox,
    const float* __restrict__ w0, const float* __restrict__ mask,
    u16* __restrict__ X0, int n){
  int i = blockIdx.x*256 + threadIdx.x;
  if (i >= n) return;
  const int R = 64, lr = 6;
  int x =  i & (R-1);
  int y = (i >> lr) & (R-1);
  int z = (i >> (2*lr)) & (R-1);
  int b =  i >> (3*lr);
  float acc[32];
  #pragma unroll
  for (int c=0;c<32;++c) acc[c] = 0.f;
  float mk = mask[i];
  if (mk != 0.f){
    for (int dz=-1; dz<=1; ++dz){
      int zz=z+dz; if ((unsigned)zz >= (unsigned)R) continue;
      for (int dy=-1; dy<=1; ++dy){
        int yy=y+dy; if ((unsigned)yy >= (unsigned)R) continue;
        for (int dx=-1; dx<=1; ++dx){
          int xx=x+dx; if ((unsigned)xx >= (unsigned)R) continue;
          float vv = vox[(size_t)((((b<<lr)+zz)<<lr)+yy)*R + xx];
          if (vv != 0.f){
            int nb = (dz+1)*9 + (dy+1)*3 + (dx+1);
            #pragma unroll
            for (int c=0;c<32;++c) acc[c] += w0[c*27+nb]*vv;
          }
        }
      }
    }
  }
  #pragma unroll
  for (int c=0;c<32;++c) X0[(size_t)i*32 + c] = f2bf(acc[c]*mk);
}

// ---------------------------------------------------------------- weight packing to B-fragment layout
// B (32x16 bf16 per chunk): lane l, element j -> k_local = (l<16?0:8)+(j&7)+((j&8)<<1), n = l&15.
// Stored as [ntile][kchunk][lane][16 halves].
__global__ __launch_bounds__(256) void k_packB(const float* __restrict__ W, u16* __restrict__ Bp,
    int Cin, int Cout, int mode, int parity, int n){
  int i = blockIdx.x*256 + threadIdx.x;
  if (i >= n) return;
  int taps = (mode==0) ? 27 : (mode==1) ? 8 : 1;
  int Kc = (taps*Cin) >> 5;
  int j    = i & 15;
  int lane = (i >> 4) & 31;
  int fc   = i >> 9;                 // ntile*Kc + kc
  int kc   = fc % Kc;
  int ntile= fc / Kc;
  int klocal = ((lane < 16) ? 0 : 8) + (j & 7) + ((j & 8) << 1);
  int k  = (kc << 5) + klocal;
  int nb = k / Cin, ci = k - nb*Cin;
  int o  = ntile*16 + (lane & 15);
  size_t widx;
  if (mode == 0)      widx = ((size_t)o*Cin + ci)*27 + nb;       // OIDHW 3x3x3
  else if (mode == 1) widx = ((size_t)o*Cin + ci)*8  + nb;       // OIDHW 2x2x2
  else                widx = ((size_t)ci*Cout + o)*8 + parity;   // IODHW 2x2x2 (deconv)
  Bp[i] = f2bf(W[widx]);
}

// ---------------------------------------------------------------- implicit-GEMM conv via WMMA (bf16 -> f32)
// MODE 0: 3x3x3 SAME, Rin==Rout.  MODE 1: 2x2x2 stride-2, Rout=Rin/2.
// MODE 2: deconv parity GEMM; M enumerates coarse (Rin) voxels, out fine voxel = 2*v + parity.
// CC = Cin/32 (compile-time). One wave computes a 32(M) x 32(N) output tile.
template<int MODE, int CC>
__global__ __launch_bounds__(32) void k_gemm_t(
    const u16* __restrict__ A, int csA,
    const u16* __restrict__ Bp,
    u16* __restrict__ O, int csO, int offO,
    const float* __restrict__ mO,
    int lrIn, int lrOut, int parity)
{
  constexpr int TAPS = (MODE==0) ? 27 : (MODE==1) ? 8 : 1;
  constexpr int KCPL = TAPS*CC;                  // K-chunks per layer
  const int lane   = threadIdx.x & 31;
  const int mt     = blockIdx.x;                 // 32-voxel M tile
  const int ntile2 = blockIdx.y;                 // 32-channel N tile
  const int lrM = (MODE==2) ? lrIn : lrOut;
  const int Rm  = 1 << lrM;
  const int Rin = 1 << lrIn;
  const int RR  = 1 << (2*lrIn), R1 = 1 << lrIn;

  int vx[2], vy[2], vz[2];
  int baseIn[2];
  #pragma unroll
  for (int s = 0; s < 2; ++s){
    int vox = mt*32 + s*16 + (lane & 15);
    int x =  vox & (Rm-1);
    int y = (vox >> lrM) & (Rm-1);
    int z = (vox >> (2*lrM)) & (Rm-1);
    int b =  vox >> (3*lrM);
    vx[s] = x; vy[s] = y; vz[s] = z;
    if (MODE == 1){ baseIn[s] = (((((b << lrIn) + 2*z) << lrIn) + 2*y) << lrIn) + 2*x; }
    else          { baseIn[s] = (((((b << lrIn) +   z) << lrIn) +   y) << lrIn) +   x; }
  }

  v8f acc[2][2];
  #pragma unroll
  for (int s = 0; s < 2; ++s)
    #pragma unroll
    for (int n = 0; n < 2; ++n) acc[s][n] = (v8f){};

  const int r1 = (lane < 16) ? 0 : 8;  // A-layout: lanes0-15 K runs {0..7,16..23}; lanes16-31 {8..15,24..31}

  const u16* bp0 = Bp + (((size_t)(2*ntile2)*KCPL) << 9) + (size_t)lane*16;
  const u16* bp1 = bp0 + ((size_t)KCPL << 9);
  const u16* ap0 = A + (size_t)baseIn[0]*csA + r1;
  const u16* ap1 = A + (size_t)baseIn[1]*csA + r1;

  int dx = -1, dy = -1, dz = -1;       // incremental 3x3x3 tap decode (MODE 0)
  for (int tap = 0; tap < TAPS; ++tap){
    int off;
    bool valid0 = true, valid1 = true;
    if (MODE == 0){
      off = dz*RR + dy*R1 + dx;
      valid0 = ((unsigned)(vz[0]+dz) < (unsigned)Rin) & ((unsigned)(vy[0]+dy) < (unsigned)Rin)
             & ((unsigned)(vx[0]+dx) < (unsigned)Rin);
      valid1 = ((unsigned)(vz[1]+dz) < (unsigned)Rin) & ((unsigned)(vy[1]+dy) < (unsigned)Rin)
             & ((unsigned)(vx[1]+dx) < (unsigned)Rin);
    } else if (MODE == 1){
      off = ((tap >> 2) << (2*lrIn)) + (((tap >> 1) & 1) << lrIn) + (tap & 1);
    } else {
      off = 0;
    }
    const u16* src0 = ap0 + (size_t)off*csA;
    const u16* src1 = ap1 + (size_t)off*csA;

    #pragma unroll
    for (int cc = 0; cc < CC; ++cc){
      Frag fa0, fa1, fb0, fb1;
      if (valid0){
        fa0.q[0] = *(const uint4*)(src0 + cc*32);
        fa0.q[1] = *(const uint4*)(src0 + cc*32 + 16);
      } else {
        fa0.q[0] = make_uint4(0u,0u,0u,0u);
        fa0.q[1] = make_uint4(0u,0u,0u,0u);
      }
      if (valid1){
        fa1.q[0] = *(const uint4*)(src1 + cc*32);
        fa1.q[1] = *(const uint4*)(src1 + cc*32 + 16);
      } else {
        fa1.q[0] = make_uint4(0u,0u,0u,0u);
        fa1.q[1] = make_uint4(0u,0u,0u,0u);
      }
      fb0.q[0] = ((const uint4*)bp0)[0];
      fb0.q[1] = ((const uint4*)bp0)[1];
      fb1.q[0] = ((const uint4*)bp1)[0];
      fb1.q[1] = ((const uint4*)bp1)[1];
      bp0 += 512; bp1 += 512;
      acc[0][0] = __builtin_amdgcn_wmma_f32_16x16x32_bf16(false, fa0.v, false, fb0.v, (short)0, acc[0][0], false, false);
      acc[0][1] = __builtin_amdgcn_wmma_f32_16x16x32_bf16(false, fa0.v, false, fb1.v, (short)0, acc[0][1], false, false);
      acc[1][0] = __builtin_amdgcn_wmma_f32_16x16x32_bf16(false, fa1.v, false, fb0.v, (short)0, acc[1][0], false, false);
      acc[1][1] = __builtin_amdgcn_wmma_f32_16x16x32_bf16(false, fa1.v, false, fb1.v, (short)0, acc[1][1], false, false);
    }
    if (MODE == 0){                    // increment (dz,dy,dx)
      if (++dx == 2){ dx = -1; if (++dy == 2){ dy = -1; ++dz; } }
    }
  }

  // C layout: VGPR rr -> M = rr + (lane<16 ? 0 : 8), N = lane&15
  int col = ntile2*32 + (lane & 15);
  #pragma unroll
  for (int s = 0; s < 2; ++s){
    int mbase = mt*32 + s*16 + ((lane < 16) ? 0 : 8);
    #pragma unroll
    for (int rr = 0; rr < 8; ++rr){
      int v2 = mbase + rr;
      size_t ovox;
      if (MODE == 2){
        int xx =  v2 & (Rm-1);
        int yy = (v2 >> lrM) & (Rm-1);
        int zz = (v2 >> (2*lrM)) & (Rm-1);
        int bb =  v2 >> (3*lrM);
        int oz = 2*zz + ((parity>>2)&1), oy = 2*yy + ((parity>>1)&1), ox = 2*xx + (parity&1);
        ovox = (size_t)(((((bb << lrOut) + oz) << lrOut) + oy) << lrOut) + ox;
      } else {
        ovox = (size_t)v2;
      }
      float m = mO[ovox];
      u16* dst = O + ovox*(size_t)csO + offO + col;
      dst[0]  = f2bf(acc[s][0][rr]*m);
      dst[16] = f2bf(acc[s][1][rr]*m);
    }
  }
}

// ---------------------------------------------------------------- sparse BN (stats over active sites)
__global__ __launch_bounds__(256) void k_bn_stats(const u16* __restrict__ X, int cs, int off,
    const float* __restrict__ mask, int nvox, int C, float* __restrict__ st){
  extern __shared__ float s[];                 // [C] sum, [C] sumsq, [1] count
  for (int t=threadIdx.x; t < 2*C+1; t += 256) s[t] = 0.f;
  __syncthreads();
  long total = (long)nvox*C;
  for (long i = (long)blockIdx.x*256 + threadIdx.x; i < total; i += (long)gridDim.x*256){
    int c = (int)(i % C);
    int v = (int)(i / C);
    float mk = mask[v];
    if (mk != 0.f){
      float x = bf2f(X[(size_t)v*cs + off + c]);
      atomicAdd(&s[c], x);
      atomicAdd(&s[C+c], x*x);
    }
    if (c == 0) atomicAdd(&s[2*C], mk);
  }
  __syncthreads();
  for (int t=threadIdx.x; t < 2*C+1; t += 256)
    if (s[t] != 0.f) atomicAdd(&st[t], s[t]);
}

__global__ __launch_bounds__(256) void k_bn_apply(const u16* __restrict__ X, int csX, int offX,
    u16* __restrict__ Y, int csY, int offY, const float* __restrict__ mask,
    const float* __restrict__ g, const float* __restrict__ bb,
    const float* __restrict__ st, int nvox, int C){
  long i = (long)blockIdx.x*256 + threadIdx.x;
  long total = (long)nvox*C;
  if (i >= total) return;
  int c = (int)(i % C);
  int v = (int)(i / C);
  float cnt  = fmaxf(st[2*C], 1.f);
  float mean = st[c]/cnt;
  float var  = st[C+c]/cnt - mean*mean;
  float sc   = g[c]*rsqrtf(var + BN_EPS);
  float x = bf2f(X[(size_t)v*csX + offX + c]);
  float y = ((x - mean)*sc + bb[c]) * mask[v];
  Y[(size_t)v*csY + offY + c] = f2bf(fmaxf(y, 0.f));
}

// ---------------------------------------------------------------- final BN + ReLU + full-grid mean pool
__global__ __launch_bounds__(256) void k_bnf_pool(const u16* __restrict__ X,
    const float* __restrict__ mask, const float* __restrict__ g, const float* __restrict__ bb,
    const float* __restrict__ st, float* __restrict__ pooled, int nvox, int R3){
  __shared__ float s[64];
  if (threadIdx.x < 64) s[threadIdx.x] = 0.f;
  __syncthreads();
  long total = (long)nvox*32;
  for (long i = (long)blockIdx.x*256 + threadIdx.x; i < total; i += (long)gridDim.x*256){
    int c = (int)(i & 31);
    int v = (int)(i >> 5);
    int b = v / R3;
    float cnt  = fmaxf(st[64], 1.f);
    float mean = st[c]/cnt;
    float var  = st[32+c]/cnt - mean*mean;
    float sc   = g[c]*rsqrtf(var + BN_EPS);
    float x = bf2f(X[(size_t)v*32 + c]);
    float y = fmaxf(((x - mean)*sc + bb[c]) * mask[v], 0.f);
    atomicAdd(&s[(b<<5)|c], y);
  }
  __syncthreads();
  if (threadIdx.x < 64 && s[threadIdx.x] != 0.f) atomicAdd(&pooled[threadIdx.x], s[threadIdx.x]);
}

__global__ __launch_bounds__(256) void k_proj(const float* __restrict__ pooled,
    const float* __restrict__ pw, const float* __restrict__ pb,
    float* __restrict__ out, float invR3){
  int i = blockIdx.x*256 + threadIdx.x;
  if (i >= 512) return;
  int b = i >> 8, j = i & 255;
  float acc = pb[j];
  #pragma unroll
  for (int c=0;c<32;++c) acc += pooled[b*32+c]*invR3*pw[c*256+j];
  out[i] = acc;
}

// ================================================================ host orchestration
extern "C" void kernel_launch(void* const* d_in, const int* in_sizes, int n_in,
                              void* d_out, int out_size, void* d_ws, size_t ws_size,
                              hipStream_t stream) {
  (void)in_sizes; (void)n_in; (void)out_size; (void)ws_size;
  auto fin = [&](int i)->const float*{ return (const float*)d_in[i]; };

  // dict flatten order: sorted keys, depth-first
  int ii = 0;
  const float* P_bnf_b = fin(ii++);
  const float* P_bnf_g = fin(ii++);
  struct LvlP { const float *ba_b,*ba_g,*bd_b,*bd_g,*bp_b,*bp_g,*bu_b,*bu_g,*wa,*wd,*wp,*wu; };
  LvlP L[4] = {};
  for (int l=0; l<4; ++l){
    L[l].ba_b = fin(ii++); L[l].ba_g = fin(ii++);
    if (l < 3){
      L[l].bd_b = fin(ii++); L[l].bd_g = fin(ii++);
      L[l].bp_b = fin(ii++); L[l].bp_g = fin(ii++);
      L[l].bu_b = fin(ii++); L[l].bu_g = fin(ii++);
      L[l].wa = fin(ii++); L[l].wd = fin(ii++); L[l].wp = fin(ii++); L[l].wu = fin(ii++);
    } else {
      L[l].wa = fin(ii++);
    }
  }
  const float* P_proj_b = fin(ii++);
  const float* P_proj_w = fin(ii++);
  const float* P_w0     = fin(ii++);
  const float* P_vox    = fin(ii++);

  const int PL[4] = {32,64,96,128};
  const int nv[4] = {2*64*64*64, 2*32*32*32, 2*16*16*16, 2*8*8*8};

  // bump allocator over workspace
  char* wsp = (char*)d_ws;
  size_t off = 0;
  auto alloc = [&](size_t bytes)->void*{
    off = (off + 255) & ~(size_t)255;
    void* p = wsp + off;
    off += bytes;
    return p;
  };

  float* mask0 = (float*)alloc(sizeof(float)*nv[0]);
  float* mask1 = (float*)alloc(sizeof(float)*nv[1]);
  float* mask2 = (float*)alloc(sizeof(float)*nv[2]);
  float* mask3 = (float*)alloc(sizeof(float)*nv[3]);

  u16* x0 = (u16*)alloc(sizeof(u16)*(size_t)nv[0]*32);
  u16* t0 = (u16*)alloc(sizeof(u16)*(size_t)nv[0]*32);
  u16* c0 = (u16*)alloc(sizeof(u16)*(size_t)nv[0]*64);
  u16* p0 = (u16*)alloc(sizeof(u16)*(size_t)nv[0]*64);
  u16* x1 = (u16*)alloc(sizeof(u16)*(size_t)nv[1]*64);
  u16* t1 = (u16*)alloc(sizeof(u16)*(size_t)nv[1]*64);
  u16* c1 = (u16*)alloc(sizeof(u16)*(size_t)nv[1]*128);
  u16* p1 = (u16*)alloc(sizeof(u16)*(size_t)nv[1]*128);
  u16* x2 = (u16*)alloc(sizeof(u16)*(size_t)nv[2]*96);
  u16* t2 = (u16*)alloc(sizeof(u16)*(size_t)nv[2]*96);
  u16* c2 = (u16*)alloc(sizeof(u16)*(size_t)nv[2]*192);
  u16* p2 = (u16*)alloc(sizeof(u16)*(size_t)nv[2]*192);
  u16* x3 = (u16*)alloc(sizeof(u16)*(size_t)nv[3]*128);
  u16* t3 = (u16*)alloc(sizeof(u16)*(size_t)nv[3]*128);
  u16* y3 = (u16*)alloc(sizeof(u16)*(size_t)nv[3]*128);

  u16* BpA[4]; u16* BpD[3]; u16* BpP[3]; u16* BpU[3];
  for (int l=0; l<4; ++l) BpA[l] = (u16*)alloc(sizeof(u16)*(size_t)PL[l]*PL[l]*27);
  for (int l=0; l<3; ++l) BpD[l] = (u16*)alloc(sizeof(u16)*(size_t)PL[l+1]*PL[l]*8);
  for (int l=0; l<3; ++l) BpP[l] = (u16*)alloc(sizeof(u16)*(size_t)PL[l]*2*PL[l]*27);
  for (int l=0; l<3; ++l) BpU[l] = (u16*)alloc(sizeof(u16)*(size_t)PL[l]*PL[l+1]*8);

  float* stats  = (float*)alloc(sizeof(float)*512);
  float* pooled = (float*)alloc(sizeof(float)*64);

  auto pack = [&](const float* W, u16* Bp, int Cin, int Cout, int mode, int parity){
    int taps = (mode==0) ? 27 : (mode==1) ? 8 : 1;
    int n = Cout*Cin*taps;
    k_packB<<<(n+255)/256, 256, 0, stream>>>(W, Bp, Cin, Cout, mode, parity, n);
  };
  auto bn = [&](const u16* X,int csX,int offX, u16* Y,int csY,int offY,
                const float* mk, const float* g, const float* b, int nvox, int C){
    k_zero<<<(2*C+1+255)/256, 256, 0, stream>>>(stats, 2*C+1);
    long total = (long)nvox*C;
    int sgrid = (int)((total + 255)/256); if (sgrid > 4096) sgrid = 4096;
    k_bn_stats<<<sgrid, 256, (2*C+1)*sizeof(float), stream>>>(X,csX,offX,mk,nvox,C,stats);
    int agrid = (int)((total + 255)/256);
    k_bn_apply<<<agrid, 256, 0, stream>>>(X,csX,offX,Y,csY,offY,mk,g,b,stats,nvox,C);
  };
  auto gemm = [&](const u16* A,int csA, const u16* Bw, u16* O,int csO,int offO,
                  const float* mO, int lrIn,int lrOut,int Cin,int Cout,int mode,int parity){
    int lrM = (mode==2) ? lrIn : lrOut;
    int msites = 2 << (3*lrM);
    dim3 g(msites/32, Cout/32, 1);
    int CC = Cin/32;
    #define GEMM_CASE(M,C) k_gemm_t<M,C><<<g,32,0,stream>>>(A,csA,Bw,O,csO,offO,mO,lrIn,lrOut,parity)
    if (mode == 0){
      switch (CC){
        case 1: GEMM_CASE(0,1); break;
        case 2: GEMM_CASE(0,2); break;
        case 3: GEMM_CASE(0,3); break;
        case 4: GEMM_CASE(0,4); break;
        default: GEMM_CASE(0,6); break;
      }
    } else if (mode == 1){
      switch (CC){
        case 1: GEMM_CASE(1,1); break;
        case 2: GEMM_CASE(1,2); break;
        default: GEMM_CASE(1,3); break;
      }
    } else {
      switch (CC){
        case 2: GEMM_CASE(2,2); break;
        case 3: GEMM_CASE(2,3); break;
        default: GEMM_CASE(2,4); break;
      }
    }
    #undef GEMM_CASE
  };

  // ---- masks
  k_mask<<<(nv[0]+255)/256, 256, 0, stream>>>(P_vox, mask0, nv[0]);
  k_pool_mask<<<(nv[1]+255)/256, 256, 0, stream>>>(mask0, mask1, 5, nv[1]);
  k_pool_mask<<<(nv[2]+255)/256, 256, 0, stream>>>(mask1, mask2, 4, nv[2]);
  k_pool_mask<<<(nv[3]+255)/256, 256, 0, stream>>>(mask2, mask3, 3, nv[3]);

  // ---- weight packing
  for (int l=0; l<4; ++l) pack(L[l].wa, BpA[l], PL[l], PL[l], 0, 0);
  for (int l=0; l<3; ++l) pack(L[l].wd, BpD[l], PL[l], PL[l+1], 1, 0);
  for (int l=0; l<3; ++l) pack(L[l].wp, BpP[l], 2*PL[l], PL[l], 0, 0);
  for (int l=0; l<3; ++l)
    for (int p=0; p<8; ++p)
      pack(L[l].wu, BpU[l] + (size_t)p*PL[l]*PL[l+1], PL[l+1], PL[l], 2, p);

  // ---- stem: 1->32 submanifold conv
  k_conv0<<<(nv[0]+255)/256, 256, 0, stream>>>(P_vox, P_w0, mask0, x0, nv[0]);

  // ---- UNet down path
  bn(x0,32,0,  t0,32,0,  mask0, L[0].ba_g, L[0].ba_b, nv[0], 32);
  gemm(t0,32, BpA[0], c0,64,0,  mask0, 6,6, 32,32, 0,0);              // conv_a0 -> skip
  bn(c0,64,0,  t0,32,0,  mask0, L[0].bd_g, L[0].bd_b, nv[0], 32);
  gemm(t0,32, BpD[0], x1,64,0,  mask1, 6,5, 32,64, 1,0);              // down 0->1

  bn(x1,64,0,  t1,64,0,  mask1, L[1].ba_g, L[1].ba_b, nv[1], 64);
  gemm(t1,64, BpA[1], c1,128,0, mask1, 5,5, 64,64, 0,0);
  bn(c1,128,0, t1,64,0,  mask1, L[1].bd_g, L[1].bd_b, nv[1], 64);
  gemm(t1,64, BpD[1], x2,96,0,  mask2, 5,4, 64,96, 1,0);              // down 1->2

  bn(x2,96,0,  t2,96,0,  mask2, L[2].ba_g, L[2].ba_b, nv[2], 96);
  gemm(t2,96, BpA[2], c2,192,0, mask2, 4,4, 96,96, 0,0);
  bn(c2,192,0, t2,96,0,  mask2, L[2].bd_g, L[2].bd_b, nv[2], 96);
  gemm(t2,96, BpD[2], x3,128,0, mask3, 4,3, 96,128, 1,0);             // down 2->3

  // ---- bottom level
  bn(x3,128,0, t3,128,0, mask3, L[3].ba_g, L[3].ba_b, nv[3], 128);
  gemm(t3,128, BpA[3], y3,128,0, mask3, 3,3, 128,128, 0,0);

  // ---- up path: level 2
  bn(y3,128,0, t3,128,0, mask3, L[2].bu_g, L[2].bu_b, nv[3], 128);
  for (int p=0; p<8; ++p)
    gemm(t3,128, BpU[2] + (size_t)p*96*128, c2,192,96, mask2, 3,4, 128,96, 2,p);
  bn(c2,192,0, p2,192,0, mask2, L[2].bp_g, L[2].bp_b, nv[2], 192);
  gemm(p2,192, BpP[2], x2,96,0, mask2, 4,4, 192,96, 0,0);             // out2 (reuses x2)

  // ---- up path: level 1
  bn(x2,96,0, t2,96,0, mask2, L[1].bu_g, L[1].bu_b, nv[2], 96);
  for (int p=0; p<8; ++p)
    gemm(t2,96, BpU[1] + (size_t)p*64*96, c1,128,64, mask1, 4,5, 96,64, 2,p);
  bn(c1,128,0, p1,128,0, mask1, L[1].bp_g, L[1].bp_b, nv[1], 128);
  gemm(p1,128, BpP[1], x1,64,0, mask1, 5,5, 128,64, 0,0);             // out1 (reuses x1)

  // ---- up path: level 0
  bn(x1,64,0, t1,64,0, mask1, L[0].bu_g, L[0].bu_b, nv[1], 64);
  for (int p=0; p<8; ++p)
    gemm(t1,64, BpU[0] + (size_t)p*32*64, c0,64,32, mask0, 5,6, 64,32, 2,p);
  bn(c0,64,0, p0,64,0, mask0, L[0].bp_g, L[0].bp_b, nv[0], 64);
  gemm(p0,64, BpP[0], x0,32,0, mask0, 6,6, 64,32, 0,0);               // out0 (reuses x0)

  // ---- final BN + ReLU + global mean pool + projection
  k_zero<<<1, 256, 0, stream>>>(stats, 512);
  {
    long total = (long)nv[0]*32;
    int sgrid = (int)((total + 255)/256); if (sgrid > 4096) sgrid = 4096;
    k_bn_stats<<<sgrid, 256, 65*sizeof(float), stream>>>(x0, 32, 0, mask0, nv[0], 32, stats);
  }
  k_zero<<<1, 256, 0, stream>>>(pooled, 64);
  k_bnf_pool<<<2048, 256, 0, stream>>>(x0, mask0, P_bnf_g, P_bnf_b, stats, pooled, nv[0], 64*64*64);
  k_proj<<<2, 256, 0, stream>>>(pooled, P_proj_w, P_proj_b, (float*)d_out, 1.f/(64.f*64.f*64.f));
}
